// FP8Linear_18451179504171
// MI455X (gfx1250) — 8-bit matrix operand order repaired and checked by reading, compiled, not run on hardware
//
#include <hip/hip_runtime.h>

#define FP8_MAX  448.0f
#define EPS_F    1e-12f
#define MARGIN   0.9f

typedef __attribute__((ext_vector_type(16))) int   v16i;
typedef __attribute__((ext_vector_type(8)))  float v8f;

#if __has_builtin(__builtin_amdgcn_global_load_async_to_lds_b128)
#define USE_ASYNC_LDS 1
typedef int v4i_ __attribute__((vector_size(16)));
typedef __attribute__((address_space(1))) v4i_ gv4i;
typedef __attribute__((address_space(3))) v4i_ lv4i;
__device__ __forceinline__ void async_b128(const void* g, void* l) {
  __builtin_amdgcn_global_load_async_to_lds_b128((gv4i*)g, (lv4i*)l, 0, 0);
}
__device__ __forceinline__ void wait_async0() {
#if __has_builtin(__builtin_amdgcn_s_wait_asynccnt)
  __builtin_amdgcn_s_wait_asynccnt(0);
#else
  asm volatile("s_wait_asynccnt 0x0" ::: "memory");
#endif
}
#else
#define USE_ASYNC_LDS 0
#endif

__device__ __forceinline__ float wave_reduce_max(float v) {
#pragma unroll
  for (int off = 16; off > 0; off >>= 1)
    v = fmaxf(v, __shfl_xor(v, off, 32));
  return v;
}

__global__ void init_amax_kernel(unsigned* p) {
  if (threadIdx.x < 2) p[threadIdx.x] = 0u;
}

__global__ void amax_kernel(const float* __restrict__ p, long n, unsigned* out) {
  long i = (long)blockIdx.x * blockDim.x + threadIdx.x;
  long stride = (long)gridDim.x * blockDim.x;
  float m = 0.0f;
  for (long j = i * 4; j < n; j += stride * 4) {
    float4 v = *(const float4*)(p + j);
    m = fmaxf(m, fmaxf(fmaxf(fabsf(v.x), fabsf(v.y)),
                       fmaxf(fabsf(v.z), fabsf(v.w))));
  }
  m = wave_reduce_max(m);
  __shared__ float red[8];
  int lane = threadIdx.x & 31, w = threadIdx.x >> 5;
  if (lane == 0) red[w] = m;
  __syncthreads();
  if (threadIdx.x == 0) {
    float t = red[0];
    for (int k = 1; k < (int)(blockDim.x >> 5); ++k) t = fmaxf(t, red[k]);
    atomicMax(out, __float_as_uint(t));
  }
}

__device__ __forceinline__ unsigned cvt_e4m3_sw(float x) {
  unsigned b = __float_as_uint(x);
  unsigned s = (b >> 24) & 0x80u;
  float a = fabsf(x);
  if (a < 0.015625f) {
    int m = (int)rintf(a * 512.0f);
    if (m >= 8) return s | 0x08u;
    return s | (unsigned)m;
  }
  unsigned ab = b & 0x7FFFFFFFu;
  unsigned lsb = (ab >> 20) & 1u;
  ab += 0x0007FFFFu + lsb;
  int e = (int)((ab >> 23) & 0xFFu) - 127 + 7;
  unsigned m = (ab >> 20) & 7u;
  if (e >= 16 || (e == 15 && m == 7)) return s | 0x7Eu;
  if (e <= 0) return s | 0x08u;
  return s | ((unsigned)e << 3) | m;
}

__device__ __forceinline__ unsigned pack4_fp8(float a, float b, float c, float d) {
  a = fminf(fmaxf(a, -FP8_MAX), FP8_MAX);
  b = fminf(fmaxf(b, -FP8_MAX), FP8_MAX);
  c = fminf(fmaxf(c, -FP8_MAX), FP8_MAX);
  d = fminf(fmaxf(d, -FP8_MAX), FP8_MAX);
#if __has_builtin(__builtin_amdgcn_cvt_pk_fp8_f32)
  int v = 0;
  v = __builtin_amdgcn_cvt_pk_fp8_f32(a, b, v, false);
  v = __builtin_amdgcn_cvt_pk_fp8_f32(c, d, v, true);
  return (unsigned)v;
#else
  return cvt_e4m3_sw(a) | (cvt_e4m3_sw(b) << 8) |
         (cvt_e4m3_sw(c) << 16) | (cvt_e4m3_sw(d) << 24);
#endif
}

__global__ void quant_kernel(const float* __restrict__ in, uint2* __restrict__ out,
                             long n, const unsigned* __restrict__ amax_bits) {
  float amax = __uint_as_float(*amax_bits);
  float s = FP8_MAX / (amax + EPS_F) * MARGIN;
  long i = (long)blockIdx.x * blockDim.x + threadIdx.x;
  long stride = (long)gridDim.x * blockDim.x;
  long n8 = n >> 3;
  for (long j = i; j < n8; j += stride) {
    float4 a = *(const float4*)(in + j * 8);
    float4 b = *(const float4*)(in + j * 8 + 4);
    uint2 r;
    r.x = pack4_fp8(a.x * s, a.y * s, a.z * s, a.w * s);
    r.y = pack4_fp8(b.x * s, b.y * s, b.z * s, b.w * s);
    out[j] = r;
  }
}

#define TILE_K     128
#define LDS_STRIDE 144

__global__ __launch_bounds__(256)
void fp8_gemm_kernel(const unsigned char* __restrict__ Aq,
                     const unsigned char* __restrict__ Bq,
                     const float* __restrict__ bias,
                     float* __restrict__ out,
                     const unsigned* __restrict__ amax2,
                     int M, int N, int K) {
#if USE_ASYNC_LDS
  __shared__ __align__(16) unsigned char sA[2][128 * LDS_STRIDE];
  __shared__ __align__(16) unsigned char sB[2][128 * LDS_STRIDE];
#else
  __shared__ __align__(16) unsigned char sA[1][128 * LDS_STRIDE];
  __shared__ __align__(16) unsigned char sB[1][128 * LDS_STRIDE];
#endif

  const int tid  = threadIdx.x;
  const int lane = tid & 31;
  const int wave = tid >> 5;
  const int wm   = wave & 3;
  const int wn   = wave >> 2;

  const int mblk = blockIdx.y * 128;
  const int nblk = blockIdx.x * 128;

  const float ax = __uint_as_float(amax2[0]);
  const float aw = __uint_as_float(amax2[1]);
  const float kInv = 1.0f / (FP8_MAX * MARGIN);
  const float inv = ((ax + EPS_F) * kInv) * ((aw + EPS_F) * kInv);

  v8f acc[2][4];
#pragma unroll
  for (int i = 0; i < 2; ++i)
#pragma unroll
    for (int j = 0; j < 4; ++j)
#pragma unroll
      for (int e = 0; e < 8; ++e) acc[i][j][e] = 0.0f;

  const int row = tid >> 1;
  const int seg = (tid & 1) * 64;
  const unsigned char* gA = Aq + (long)(mblk + row) * K + seg;
  const unsigned char* gB = Bq + (long)(nblk + row) * K + seg;
  const int lds_off = row * LDS_STRIDE + seg;

  const int mlane  = lane & 15;
  const int aoff_k = (lane >> 4) * 8;
  const int boff_k = (lane >> 4) * 8;

  const int nk = K / TILE_K;

#if USE_ASYNC_LDS
#pragma unroll
  for (int i = 0; i < 4; ++i) {
    async_b128(gA + i * 16, sA[0] + lds_off + i * 16);
    async_b128(gB + i * 16, sB[0] + lds_off + i * 16);
  }
  wait_async0();
  __syncthreads();
#endif

  for (int kt = 0; kt < nk; ++kt) {
#if USE_ASYNC_LDS
    const int cur = kt & 1;
    if (kt + 1 < nk) {
      const int knext = (kt + 1) * TILE_K;
      const int nxt = 1 - cur;
#pragma unroll
      for (int i = 0; i < 4; ++i) {
        async_b128(gA + knext + i * 16, sA[nxt] + lds_off + i * 16);
        async_b128(gB + knext + i * 16, sB[nxt] + lds_off + i * 16);
      }
    }
#else
    const int cur = 0;
    const int k0 = kt * TILE_K;
    __syncthreads();
#pragma unroll
    for (int i = 0; i < 4; ++i) {
      *(uint4*)(sA[0] + lds_off + i * 16) = *(const uint4*)(gA + k0 + i * 16);
      *(uint4*)(sB[0] + lds_off + i * 16) = *(const uint4*)(gB + k0 + i * 16);
    }
    __syncthreads();
#endif

    v16i afrag[2], bfrag[4];
#pragma unroll
    for (int i = 0; i < 2; ++i) {
      const unsigned char* base =
          sA[cur] + (wm * 32 + i * 16 + mlane) * LDS_STRIDE + aoff_k;
#pragma unroll
      for (int h = 0; h < 2; ++h)
#pragma unroll
        for (int p = 0; p < 4; ++p) {
          uint2 t = *(const uint2*)(base + h * 64 + p * 16);
          afrag[i][h * 8 + p * 2]     = (int)t.x;
          afrag[i][h * 8 + p * 2 + 1] = (int)t.y;
        }
    }
#pragma unroll
    for (int j = 0; j < 4; ++j) {
      const unsigned char* base =
          sB[cur] + (wn * 64 + j * 16 + mlane) * LDS_STRIDE + boff_k;
#pragma unroll
      for (int g = 0; g < 4; ++g) {
        uint2 t = *(const uint2*)(base + g * 32);
        uint2 u = *(const uint2*)(base + g * 32 + 16);
        bfrag[j][4 * g]     = (int)t.x;
        bfrag[j][4 * g + 1] = (int)t.y;
        bfrag[j][4 * g + 2] = (int)u.x;
        bfrag[j][4 * g + 3] = (int)u.y;
      }
    }
#pragma unroll
    for (int i = 0; i < 2; ++i)
#pragma unroll
      for (int j = 0; j < 4; ++j)
        acc[i][j] = __builtin_amdgcn_wmma_f32_16x16x128_fp8_fp8(
            afrag[i], bfrag[j], (short)0, acc[i][j], false, false);

#if USE_ASYNC_LDS
    if (kt + 1 < nk) wait_async0();
    __syncthreads();
#endif
  }

  const int mrow0 = mblk + wm * 32 + (lane >> 4) * 8;
#pragma unroll
  for (int j = 0; j < 4; ++j) {
    int n = nblk + wn * 64 + j * 16 + mlane;
    float bv = bias[n];
#pragma unroll
    for (int i = 0; i < 2; ++i) {
      int mb = mrow0 + i * 16;
#pragma unroll
      for (int r = 0; r < 8; ++r)
        out[(long)(mb + r) * N + n] = acc[i][j][r] * inv + bv;
    }
  }
}

extern "C" void kernel_launch(void* const* d_in, const int* in_sizes, int n_in,
                              void* d_out, int out_size, void* d_ws, size_t ws_size,
                              hipStream_t stream) {
  const float* x    = (const float*)d_in[0];
  const float* w    = (const float*)d_in[1];
  const float* bias = (const float*)d_in[2];
  float* out = (float*)d_out;

  const long nx = (long)in_sizes[0];
  const long nw = (long)in_sizes[1];
  const int  N  = in_sizes[2];
  const int  K  = (int)(nw / N);
  const int  M  = (int)(nx / K);

  unsigned*      amax2 = (unsigned*)d_ws;
  unsigned char* xq    = (unsigned char*)d_ws + 256;
  unsigned char* wq    = xq + nx;

  init_amax_kernel<<<1, 32, 0, stream>>>(amax2);
  amax_kernel<<<1024, 256, 0, stream>>>(x, nx, amax2 + 0);
  amax_kernel<<<256,  256, 0, stream>>>(w, nw, amax2 + 1);
  quant_kernel<<<2048, 256, 0, stream>>>(x, (uint2*)xq, nx, amax2 + 0);
  quant_kernel<<<512,  256, 0, stream>>>(w, (uint2*)wq, nw, amax2 + 1);

  dim3 grid(N / 128, M / 128);
  fp8_gemm_kernel<<<grid, 256, 0, stream>>>(xq, wq, bias, out, amax2, M, N, K);
}
